// SpHopAttentionLayer_62706522522387
// MI455X (gfx1250) — matrix operand order repaired and checked by reading, not compiled, not run on hardware
//
#include <hip/hip_runtime.h>
#include <math.h>

#define N_NODES 50000
#define N_EDGES 800000
#define F_IN    256
#define F_OUT   128
#define ALPHA   0.2f

#define TM      128                 // M rows per block
#define TK      32                  // K step (f16 WMMA K)
#define NPAD    50048               // 391 * 128, zero-padded rows of xh
#define LDA     40                  // A tile row stride in halves (80 B)
#define LDBF    264                 // full-B row stride in halves (528 B)

typedef __attribute__((ext_vector_type(16))) _Float16 v16h;
typedef __attribute__((ext_vector_type(8)))  _Float16 v8h;
typedef __attribute__((ext_vector_type(4)))  _Float16 v4h;
typedef __attribute__((ext_vector_type(8)))  float    v8f;
typedef __attribute__((ext_vector_type(4)))  float    v4f;

// ---------------------------------------------------------------------------
// init: zero output accumulator, seed rowsum with the reference's 1e-8 epsilon
// ---------------------------------------------------------------------------
__global__ __launch_bounds__(256) void gat_init_kernel(float* __restrict__ out,
                                                       float* __restrict__ rowsum) {
    int i = blockIdx.x * blockDim.x + threadIdx.x;
    if (i < N_NODES * F_OUT) out[i] = 0.0f;
    if (i < N_NODES) rowsum[i] = 1e-8f;
}

// ---------------------------------------------------------------------------
// setup: asum = a.sum(axis=0)  [256];  Wt[n][k] = (f16) W[k][n]  (B^T, f16)
// ---------------------------------------------------------------------------
__global__ __launch_bounds__(256) void gat_setup_kernel(const float* __restrict__ a,
                                                        const float* __restrict__ W,
                                                        float* __restrict__ asum,
                                                        _Float16* __restrict__ Wt) {
    int tid = blockIdx.x * blockDim.x + threadIdx.x;
    if (blockIdx.x == 0) {
        int j = threadIdx.x;  // 0..255
        float s = 0.0f;
        for (int i = 0; i < 2 * F_OUT; ++i) s += a[i * (2 * F_OUT) + j];
        asum[j] = s;
    }
    for (int idx = tid; idx < F_OUT * F_IN; idx += gridDim.x * blockDim.x) {
        int n = idx / F_IN;
        int k = idx - n * F_IN;
        Wt[n * F_IN + k] = (_Float16)W[k * F_OUT + n];
    }
}

// ---------------------------------------------------------------------------
// convert: xh = (f16) x, zero-padded to NPAD rows so the GEMM's async
// LDS copies never need predication (EXEC stays all-1s for WMMA blocks).
// ---------------------------------------------------------------------------
__global__ __launch_bounds__(256) void gat_convert_kernel(const float* __restrict__ x,
                                                          _Float16* __restrict__ xh) {
    int i4 = blockIdx.x * blockDim.x + threadIdx.x;     // float4 slot
    if (i4 >= (NPAD * F_IN) / 4) return;
    int row = (i4 * 4) / F_IN;
    v4f v = {0.0f, 0.0f, 0.0f, 0.0f};
    if (row < N_NODES) v = *(const v4f*)(x + (size_t)i4 * 4);
    v4h o;
    o.x = (_Float16)v.x; o.y = (_Float16)v.y;
    o.z = (_Float16)v.z; o.w = (_Float16)v.w;
    *(v4h*)(xh + (size_t)i4 * 4) = o;
}

// ---------------------------------------------------------------------------
// GEMM: h = x @ W via v_wmma_f32_16x16x32_f16.
//  - whole B^T (128x256 f16) staged to LDS once via GLOBAL_LOAD_ASYNC_TO_LDS
//  - A tiles (128x32 f16) double-buffered, async-copied memory->LDS
//  - 8 waves, wave w owns rows [16w,16w+16) x all 128 cols (8 v8f accums)
// ---------------------------------------------------------------------------
__global__ __launch_bounds__(256) void gat_gemm_kernel(const _Float16* __restrict__ xh,
                                                       const _Float16* __restrict__ Wt,
                                                       float* __restrict__ h) {
    __shared__ _Float16 sB[F_OUT * LDBF];     // 67,584 B: full B^T, resident
    __shared__ _Float16 sA[2][TM * LDA];      // 2 x 10,240 B: A double buffer

    const int tid  = threadIdx.x;
    const int wave = tid >> 5;
    const int lane = tid & 31;
    const int l15  = lane & 15;
    const int row0 = blockIdx.x * TM;

    // ---- one-time B^T stage: 4096 x 16B async chunks (16 per thread) ----
    #pragma unroll
    for (int it = 0; it < 16; ++it) {
        int e = tid + it * 256;               // chunk 0..4095
        int n = e >> 5;                       // 32 chunks per row
        int c = (e & 31) * 8;                 // half offset
        unsigned lds = (unsigned)(uintptr_t)&sB[n * LDBF + c];
        unsigned long long g = (unsigned long long)(uintptr_t)(Wt + n * F_IN + c);
        asm volatile("global_load_async_to_lds_b128 %0, %1, off"
                     :: "v"(lds), "v"(g) : "memory");
    }

    // ---- A tile async stage: 512 x 16B chunks (2 per thread) ----
    auto stageA = [&](int k0, int b) {
        #pragma unroll
        for (int it = 0; it < 2; ++it) {
            int e = tid + it * 256;           // chunk 0..511
            int r = e >> 2;                   // 4 chunks per row
            int c = (e & 3) * 8;
            unsigned lds = (unsigned)(uintptr_t)&sA[b][r * LDA + c];
            unsigned long long g =
                (unsigned long long)(uintptr_t)(xh + (size_t)(row0 + r) * F_IN + k0 + c);
            asm volatile("global_load_async_to_lds_b128 %0, %1, off"
                         :: "v"(lds), "v"(g) : "memory");
        }
    };

    stageA(0, 0);
    asm volatile("s_wait_asynccnt 0" ::: "memory");
    __syncthreads();

    v8f acc[8] = {};
    int buf = 0;

    for (int k0 = 0; k0 < F_IN; k0 += TK) {
        if (k0 + TK < F_IN) stageA(k0 + TK, buf ^ 1);   // prefetch next tile

        // A fragment (16x32 f16 layout):
        // lanes 0-15: K {0..7,16..23}; lanes 16-31: K {8..15,24..31}
        const int arow = wave * 16 + l15;
        const int kofs = (lane < 16) ? 0 : 8;
        union { v16h v; v8h h2[2]; } au;
        au.h2[0] = *(const v8h*)&sA[buf][arow * LDA + kofs];
        au.h2[1] = *(const v8h*)&sA[buf][arow * LDA + 16 + kofs];

        // B fragment: column of B = contiguous run of sB row
        const int bk = (lane < 16) ? 0 : 8;
        #pragma unroll
        for (int j = 0; j < 8; ++j) {
            int bcol = j * 16 + l15;
            union { v16h v; v8h h2[2]; } bu;
            bu.h2[0] = *(const v8h*)&sB[bcol * LDBF + k0 + bk];
            bu.h2[1] = *(const v8h*)&sB[bcol * LDBF + k0 + bk + 16];
            acc[j] = __builtin_amdgcn_wmma_f32_16x16x32_f16(
                false, au.v, false, bu.v, (short)0, acc[j], false, false);
        }

        asm volatile("s_wait_asynccnt 0" ::: "memory");  // next A tile landed
        __syncthreads();                                 // all waves done with cur
        buf ^= 1;
    }

    // ---- store C: VGPR r -> M = (lane<16 ? r : 8+r), N = 16j + lane%16 ----
    const int mbase = row0 + wave * 16 + ((lane < 16) ? 0 : 8);
    #pragma unroll
    for (int j = 0; j < 8; ++j) {
        int col = j * 16 + l15;
        #pragma unroll
        for (int r = 0; r < 8; ++r) {
            int grow = mbase + r;
            if (grow < N_NODES) h[(size_t)grow * F_OUT + col] = acc[j][r];
        }
    }
}

// ---------------------------------------------------------------------------
// Edge phase: one wave32 per edge. Lanes each own one float4 (128/32 = 4 f32)
// of h[src] and h[dst]; shuffle-reduce for the two dots; scatter-add.
// h (25.6 MB) is L2-resident, so random row gathers resolve at L2.
// ---------------------------------------------------------------------------
__global__ __launch_bounds__(256) void gat_edge_kernel(const int* __restrict__ src,
                                                       const int* __restrict__ dst,
                                                       const float* __restrict__ h,
                                                       const float* __restrict__ asum,
                                                       float* __restrict__ out,
                                                       float* __restrict__ rowsum) {
    int gw   = (blockIdx.x * blockDim.x + threadIdx.x) >> 5;
    int lane = threadIdx.x & 31;
    if (gw >= N_EDGES) return;

    int s = src[gw];
    int d = dst[gw];

    v4f vs = *(const v4f*)(h + (size_t)s * F_OUT + lane * 4);
    v4f vd = *(const v4f*)(h + (size_t)d * F_OUT + lane * 4);
    v4f a1 = *(const v4f*)(asum + lane * 4);           // asum[:128]
    v4f a2 = *(const v4f*)(asum + F_OUT + lane * 4);   // asum[128:]

    float lin  = vs.x * a1.x + vs.y * a1.y + vs.z * a1.z + vs.w * a1.w
               + vd.x * a2.x + vd.y * a2.y + vd.z * a2.z + vd.w * a2.w;
    float conv = vs.x * vd.x + vs.y * vd.y + vs.z * vd.z + vs.w * vd.w;

    #pragma unroll
    for (int off = 16; off > 0; off >>= 1) {
        lin  += __shfl_xor(lin,  off, 32);
        conv += __shfl_xor(conv, off, 32);
    }

    float sig = 1.0f / (1.0f + __expf(-conv));
    float z   = lin * sig;
    float lz  = (z >= 0.0f) ? z : ALPHA * z;
    float ee  = __expf(-lz);

    float* o = out + (size_t)s * F_OUT + lane * 4;
    atomicAdd(o + 0, ee * vd.x);
    atomicAdd(o + 1, ee * vd.y);
    atomicAdd(o + 2, ee * vd.z);
    atomicAdd(o + 3, ee * vd.w);
    if (lane == 0) atomicAdd(rowsum + s, ee);
}

// ---------------------------------------------------------------------------
// finalize: out = elu(out / rowsum)
// ---------------------------------------------------------------------------
__global__ __launch_bounds__(256) void gat_final_kernel(float* __restrict__ out,
                                                        const float* __restrict__ rowsum) {
    int i = blockIdx.x * blockDim.x + threadIdx.x;
    if (i >= N_NODES * F_OUT) return;
    float v = out[i] / rowsum[i / F_OUT];
    out[i] = (v > 0.0f) ? v : (__expf(v) - 1.0f);
}

// ---------------------------------------------------------------------------
extern "C" void kernel_launch(void* const* d_in, const int* in_sizes, int n_in,
                              void* d_out, int out_size, void* d_ws, size_t ws_size,
                              hipStream_t stream) {
    const float* x    = (const float*)d_in[0];
    const int*   ei   = (const int*)d_in[1];     // [2, E] flat
    const float* W    = (const float*)d_in[2];
    const float* a    = (const float*)d_in[3];
    const int*   src  = ei;
    const int*   dst  = ei + N_EDGES;
    float*       out  = (float*)d_out;

    // workspace layout (all offsets 256B-aligned)
    char* ws = (char*)d_ws;
    size_t off = 0;
    float*    h      = (float*)(ws + off);  off += (size_t)N_NODES * F_OUT * 4;  // 25,600,000
    _Float16* xh     = (_Float16*)(ws + off); off += (size_t)NPAD * F_IN * 2;    // 25,624,576
    float*    asum   = (float*)(ws + off);  off += 1024;
    _Float16* Wt     = (_Float16*)(ws + off); off += (size_t)F_OUT * F_IN * 2;   // 65,536
    float*    rowsum = (float*)(ws + off);

    (void)in_sizes; (void)n_in; (void)out_size; (void)ws_size;

    const int threads = 256;
    int initBlocks = (N_NODES * F_OUT + threads - 1) / threads;      // 25000
    gat_init_kernel<<<initBlocks, threads, 0, stream>>>(out, rowsum);

    gat_setup_kernel<<<64, threads, 0, stream>>>(a, W, asum, Wt);

    int convBlocks = ((NPAD * F_IN) / 4 + threads - 1) / threads;    // 12512
    gat_convert_kernel<<<convBlocks, threads, 0, stream>>>(x, xh);

    int gemmBlocks = NPAD / TM;                                      // 391
    gat_gemm_kernel<<<gemmBlocks, threads, 0, stream>>>(xh, Wt, h);

    int edgeBlocks = (N_EDGES * 32 + threads - 1) / threads;         // 100000
    gat_edge_kernel<<<edgeBlocks, threads, 0, stream>>>(src, dst, h, asum, out, rowsum);

    gat_final_kernel<<<initBlocks, threads, 0, stream>>>(out, rowsum);
}
